// VQVAE_77721728189145
// MI455X (gfx1250) — compile-verified, matrix operand order checked by reading, not run on hardware
//
#include <hip/hip_runtime.h>
#include <hip/hip_bf16.h>

typedef _Float16 h16v __attribute__((ext_vector_type(16)));
typedef _Float16 h8v  __attribute__((ext_vector_type(8)));
typedef float    f8v  __attribute__((ext_vector_type(8)));

struct Taps16 { int dy[16]; int dx[16]; };
struct Offs16 { int off[16]; };

// ---------------------------------------------------------------------------
// Weight repack: build the per-lane WMMA B-operand image.
// Packed layout: pack[(((t*2 + c)*4 + nt)*32 + lane)*16 + e]
//   lane: n = lane&15 (cout within 16-tile), hi = lane>>4
//   e<8  -> K = hi*8 + e          (chunk-local cin, VGPRs 0..3)
//   e>=8 -> K = 16 + hi*8 + (e-8) (VGPRs 4..7)
//   cin = c*32 + K ; cout = nt*16 + n
// Source weight element: w[cout*coutStride + cin*cinStride + off[t]]
// ---------------------------------------------------------------------------
__global__ void k_repack(const float* __restrict__ w, _Float16* __restrict__ pack,
                         int ntaps, int coutStride, int cinStride, Offs16 offs) {
    int tid = blockIdx.x * blockDim.x + threadIdx.x;
    int total = ntaps * 2 * 4 * 32;
    if (tid >= total) return;
    int lane = tid & 31;
    int nt   = (tid >> 5) & 3;
    int c    = (tid >> 7) & 1;
    int t    = tid >> 8;
    int n = lane & 15, hi = lane >> 4;
    int cout = nt * 16 + n;
    _Float16* dst = pack + (size_t)tid * 16;
    #pragma unroll
    for (int e = 0; e < 16; ++e) {
        int K = (e < 8) ? (hi * 8 + e) : (16 + hi * 8 + (e - 8));
        int cin = c * 32 + K;
        dst[e] = (_Float16)w[(size_t)cout * coutStride + (size_t)cin * cinStride + offs.off[t]];
    }
}

// ---------------------------------------------------------------------------
// Generic WMMA implicit-GEMM conv, Cin=Cout=64, channels-last f16.
// One wave -> 16 pixels (a row segment of the output subgrid) x 64 couts.
// NTAPS is a compile-time constant -> fully unrolled straight-line WMMA chain
// (NTAPS*2 chunks x 4 cout-tiles). All subgrid dims are powers of two, so the
// tile decomposition is pure shift/mask (no integer division).
// Output mapping: (y,x) of subgrid -> (y*ostep+oy0, x*ostep+ox0) of full out.
// ---------------------------------------------------------------------------
template <int NTAPS>
__global__ void __launch_bounds__(256)
k_conv_wmma(const _Float16* __restrict__ in, int Hin, int Win,
            const _Float16* __restrict__ wpack, const float* __restrict__ bias,
            _Float16* __restrict__ out, int Hout, int Wout,
            int wShift, int hShift, int ostep, int oy0, int ox0,
            int istride, Taps16 taps, int relu) {
    int lane = threadIdx.x & 31;
    long wave = (long)blockIdx.x * (blockDim.x >> 5) + (threadIdx.x >> 5);
    long p0 = wave << 4;                       // 16 pixels per wave
    int x0 = (int)(p0 & ((1 << wShift) - 1));  // Ws = 1<<wShift (multiple of 16)
    long t1 = p0 >> wShift;
    int y  = (int)(t1 & ((1 << hShift) - 1));
    int b  = (int)(t1 >> hShift);

    int m = lane & 15, hi = lane >> 4;
    int x = x0 + m;

    f8v acc0 = {}, acc1 = {}, acc2 = {}, acc3 = {};

    #pragma unroll
    for (int t = 0; t < NTAPS; ++t) {
        int iy = y * istride + taps.dy[t];
        if (iy >= 0 && iy < Hin) {                 // wave-uniform (zero pad row)
            int ix = x * istride + taps.dx[t];
            bool ok = (ix >= 0) && (ix < Win);
            const _Float16* prow = in + (((size_t)b * Hin + iy) * Win + ix) * 64;
            // prefetch next tap's row to hide the stride-2 gather latency
            if (t + 1 < NTAPS) {
                int iy2 = y * istride + taps.dy[t + 1];
                int ix2 = x * istride + taps.dx[t + 1];
                if (iy2 >= 0 && iy2 < Hin && ix2 >= 0 && ix2 < Win)
                    __builtin_prefetch(in + (((size_t)b * Hin + iy2) * Win + ix2) * 64, 0, 3);
            }
            #pragma unroll
            for (int c = 0; c < 2; ++c) {
                // A operand: lane holds cin = c*32 + {hi*8..+7, 16+hi*8..+7}
                union { h16v v; struct { h8v lo; h8v hi; } p; } au;
                au.v = (h16v)(_Float16)0.f;
                if (ok) {                           // EXEC-masked b128 pair
                    const h8v* q = (const h8v*)(prow + c * 32 + hi * 8);
                    au.p.lo = q[0];
                    au.p.hi = q[2];                 // +16 halves
                }
                h16v a = au.v;
                // B operands: prepacked, one v16h per (tap,chunk,ntile,lane)
                const h16v* wq = (const h16v*)wpack + ((size_t)(t * 2 + c) * 4) * 32 + lane;
                h16v b0 = wq[0];
                acc0 = __builtin_amdgcn_wmma_f32_16x16x32_f16(false, a, false, b0, (short)0, acc0, false, false);
                h16v b1 = wq[32];
                acc1 = __builtin_amdgcn_wmma_f32_16x16x32_f16(false, a, false, b1, (short)0, acc1, false, false);
                h16v b2 = wq[64];
                acc2 = __builtin_amdgcn_wmma_f32_16x16x32_f16(false, a, false, b2, (short)0, acc2, false, false);
                h16v b3 = wq[96];
                acc3 = __builtin_amdgcn_wmma_f32_16x16x32_f16(false, a, false, b3, (short)0, acc3, false, false);
            }
        }
    }

    // D layout: element (M = r + 8*hi, N = lane&15) in acc[r]
    int n = lane & 15;
    float bb0 = bias[n], bb1 = bias[16 + n], bb2 = bias[32 + n], bb3 = bias[48 + n];
    size_t orow = ((size_t)b * Hout + (size_t)(y * ostep + oy0)) * Wout;
    #pragma unroll
    for (int r = 0; r < 8; ++r) {
        int mm = r + 8 * hi;
        size_t o = (orow + (size_t)((x0 + mm) * ostep + ox0)) * 64;
        float v0 = acc0[r] + bb0, v1 = acc1[r] + bb1, v2 = acc2[r] + bb2, v3 = acc3[r] + bb3;
        if (relu) { v0 = fmaxf(v0, 0.f); v1 = fmaxf(v1, 0.f); v2 = fmaxf(v2, 0.f); v3 = fmaxf(v3, 0.f); }
        out[o + n]      = (_Float16)v0;
        out[o + 16 + n] = (_Float16)v1;
        out[o + 32 + n] = (_Float16)v2;
        out[o + 48 + n] = (_Float16)v3;
    }
}

// ---------------------------------------------------------------------------
// Encoder layer 1: Cin=1 -> 64, 4x4 s2 p1, f32 NCHW in -> f16 channels-last out, ReLU
// ---------------------------------------------------------------------------
__global__ void __launch_bounds__(256)
k_enc1(const float* __restrict__ x, const float* __restrict__ w,
       const float* __restrict__ bias, _Float16* __restrict__ out) {
    __shared__ float ws_[64 * 16];
    __shared__ float bs_[64];
    for (int i = threadIdx.x; i < 64 * 16; i += blockDim.x) ws_[i] = w[i];
    for (int i = threadIdx.x; i < 64; i += blockDim.x) bs_[i] = bias[i];
    __syncthreads();
    long pid = (long)blockIdx.x * blockDim.x + threadIdx.x;   // [8,256,256]
    int ox = (int)(pid & 255); long t = pid >> 8;
    int oy = (int)(t & 255);   int b = (int)(t >> 8);
    float patch[16];
    #pragma unroll
    for (int ky = 0; ky < 4; ++ky)
        #pragma unroll
        for (int kx = 0; kx < 4; ++kx) {
            int iy = 2 * oy - 1 + ky, ix = 2 * ox - 1 + kx;
            bool ok = (iy >= 0) & (iy < 512) & (ix >= 0) & (ix < 512);
            patch[ky * 4 + kx] = ok ? x[((size_t)b * 512 + iy) * 512 + ix] : 0.f;
        }
    _Float16* o = out + (size_t)pid * 64;
    for (int co = 0; co < 64; ++co) {
        float s = bs_[co];
        #pragma unroll
        for (int k = 0; k < 16; ++k) s = fmaf(ws_[co * 16 + k], patch[k], s);
        o[co] = (_Float16)fmaxf(s, 0.f);
    }
}

// ---------------------------------------------------------------------------
// Vector quantizer: argmin over 512 codes, write z_q (f16), accumulate SSE.
// Codebook staged to LDS as f16 (64 KB); LDS reused for the block reduction.
// ---------------------------------------------------------------------------
__global__ void __launch_bounds__(256)
k_vq(const _Float16* __restrict__ z, const float* __restrict__ cb,
     _Float16* __restrict__ zq, float* __restrict__ sse_accum) {
    __shared__ __align__(16) _Float16 cbs[512 * 64];
    for (int i = threadIdx.x; i < 512 * 64; i += blockDim.x) cbs[i] = (_Float16)cb[i];
    __syncthreads();
    long pid = (long)blockIdx.x * blockDim.x + threadIdx.x;   // 32768 pixels
    float zv[64];
    const h8v* zp = (const h8v*)(z + (size_t)pid * 64);
    #pragma unroll
    for (int j = 0; j < 8; ++j) {
        h8v v = zp[j];
        #pragma unroll
        for (int e = 0; e < 8; ++e) zv[j * 8 + e] = (float)v[e];
    }
    float best = 3.4e38f; int bi = 0;
    for (int k = 0; k < 512; ++k) {
        const h8v* cp = (const h8v*)(cbs + k * 64);
        float d = 0.f;
        #pragma unroll
        for (int j = 0; j < 8; ++j) {
            h8v v = cp[j];
            #pragma unroll
            for (int e = 0; e < 8; ++e) { float df = zv[j * 8 + e] - (float)v[e]; d = fmaf(df, df, d); }
        }
        if (d < best) { best = d; bi = k; }
    }
    // straight-through forward value is exactly codebook[bi]
    const h8v* cp = (const h8v*)(cbs + bi * 64);
    h8v* oq = (h8v*)(zq + (size_t)pid * 64);
    #pragma unroll
    for (int j = 0; j < 8; ++j) oq[j] = cp[j];
    __syncthreads();
    float* red = (float*)cbs;            // reuse LDS for reduction
    red[threadIdx.x] = best;             // best == per-pixel SSE of chosen code
    __syncthreads();
    for (int s = 128; s > 0; s >>= 1) {
        if ((int)threadIdx.x < s) red[threadIdx.x] += red[threadIdx.x + s];
        __syncthreads();
    }
    if (threadIdx.x == 0) atomicAdd(sse_accum, red[0]);
}

// ---------------------------------------------------------------------------
// Final conv-transpose: 64 -> 1, k4 s2 p1, + sigmoid, f32 NCHW output.
// ---------------------------------------------------------------------------
__global__ void __launch_bounds__(256)
k_dect3(const _Float16* __restrict__ in, const float* __restrict__ w,
        const float* __restrict__ bias, float* __restrict__ out) {
    __shared__ float ws_[64 * 16];       // w[cin][0][ky][kx]
    for (int i = threadIdx.x; i < 1024; i += blockDim.x) ws_[i] = w[i];
    __syncthreads();
    long pid = (long)blockIdx.x * blockDim.x + threadIdx.x;   // [8,512,512]
    int ox = (int)(pid & 511); long t = pid >> 9;
    int oy = (int)(t & 511);   int b = (int)(t >> 9);
    int py = oy & 1, px = ox & 1, yb = oy >> 1, xb = ox >> 1;
    float s = bias[0];
    #pragma unroll
    for (int a2 = 0; a2 < 2; ++a2)
        #pragma unroll
        for (int a3 = 0; a3 < 2; ++a3) {
            int ky = py ? (2 * a2) : (1 + 2 * a2);
            int iy = py ? (yb + 1 - a2) : (yb - a2);
            int kx = px ? (2 * a3) : (1 + 2 * a3);
            int ix = px ? (xb + 1 - a3) : (xb - a3);
            if (iy < 0 || iy >= 256 || ix < 0 || ix >= 256) continue;
            const h8v* p = (const h8v*)(in + (((size_t)b * 256 + iy) * 256 + ix) * 64);
            int wof = ky * 4 + kx;
            #pragma unroll
            for (int j = 0; j < 8; ++j) {
                h8v v = p[j];
                #pragma unroll
                for (int e = 0; e < 8; ++e) s = fmaf((float)v[e], ws_[(j * 8 + e) * 16 + wof], s);
            }
        }
    out[pid] = 1.f / (1.f + expf(-s));
}

__global__ void k_zero(float* p) { if (blockIdx.x == 0 && threadIdx.x == 0) p[0] = 0.f; }

__global__ void k_loss(const float* __restrict__ sse, float* __restrict__ out) {
    if (blockIdx.x == 0 && threadIdx.x == 0)
        out[0] = 1.25f * sse[0] / 2097152.0f;   // mean over 8*64*64*64 elements, x1.25
}

// ---------------------------------------------------------------------------
extern "C" void kernel_launch(void* const* d_in, const int* in_sizes, int n_in,
                              void* d_out, int out_size, void* d_ws, size_t ws_size,
                              hipStream_t stream) {
    const float* x       = (const float*)d_in[0];
    const float* enc_w1  = (const float*)d_in[1];  const float* enc_b1 = (const float*)d_in[2];
    const float* enc_w2  = (const float*)d_in[3];  const float* enc_b2 = (const float*)d_in[4];
    const float* enc_w3  = (const float*)d_in[5];  const float* enc_b3 = (const float*)d_in[6];
    const float* enc_w4  = (const float*)d_in[7];  const float* enc_b4 = (const float*)d_in[8];
    const float* codebook= (const float*)d_in[9];
    const float* dec_w1  = (const float*)d_in[10]; const float* dec_b1 = (const float*)d_in[11];
    const float* dec_tw1 = (const float*)d_in[12]; const float* dec_tb1= (const float*)d_in[13];
    const float* dec_tw2 = (const float*)d_in[14]; const float* dec_tb2= (const float*)d_in[15];
    const float* dec_tw3 = (const float*)d_in[16]; const float* dec_tb3= (const float*)d_in[17];

    char* ws = (char*)d_ws;
    size_t off = 0;
    auto alloc = [&](size_t bytes) { size_t o = off; off = (off + bytes + 255) & ~(size_t)255; return o; };
    size_t oA1 = alloc(67108864);   // a1 [8,256,256,64] f16 ; later reused as DT2
    size_t oA2 = alloc(16777216);   // a2 [8,128,128,64] f16 ; later reused as DT1
    size_t oA3 = alloc(4194304);    // a3 [8,64,64,64]  f16 ; later reused as ZQ
    size_t oZ  = alloc(4194304);    // z               f16 ; later reused as D1
    size_t oPe2 = alloc(131072);    // packed enc_w2
    size_t oPe3 = alloc(131072);    // packed enc_w3
    size_t oPe4 = alloc(73728);     // packed enc_w4 (3x3)
    size_t oPd1 = alloc(73728);     // packed dec_w1 (3x3)
    size_t oPt1 = alloc(131072);    // packed dec_tw1, 4 parities x 32768 B
    size_t oPt2 = alloc(131072);    // packed dec_tw2
    size_t oAcc = alloc(256);
    (void)ws_size; (void)in_sizes; (void)n_in; (void)out_size;

    _Float16* A1  = (_Float16*)(ws + oA1);
    _Float16* A2  = (_Float16*)(ws + oA2);
    _Float16* A3  = (_Float16*)(ws + oA3);
    _Float16* Z   = (_Float16*)(ws + oZ);
    _Float16* ZQ  = A3;                     // reuse
    _Float16* D1  = Z;                      // reuse
    _Float16* DT1 = A2;                     // reuse
    _Float16* DT2 = A1;                     // reuse
    float* acc = (float*)(ws + oAcc);

    // ---- taps/offsets for dense convs ----
    Taps16 tp4{}; Offs16 of4{};             // 4x4 (enc s2 convs), dy/dx = k-1
    for (int t = 0; t < 16; ++t) { of4.off[t] = t; tp4.dy[t] = t / 4 - 1; tp4.dx[t] = t % 4 - 1; }
    Taps16 tp3{}; Offs16 of3{};             // 3x3 s1 p1
    for (int t = 0; t < 9; ++t) { of3.off[t] = t; tp3.dy[t] = t / 3 - 1; tp3.dx[t] = t % 3 - 1; }

    // ---- weight repacking (B-operand register images) ----
    // OIHW [64,64,kh,kw]: coutStride = 64*kh*kw, cinStride = kh*kw
    k_repack<<<16, 256, 0, stream>>>(enc_w2, (_Float16*)(ws + oPe2), 16, 1024, 16, of4);
    k_repack<<<16, 256, 0, stream>>>(enc_w3, (_Float16*)(ws + oPe3), 16, 1024, 16, of4);
    k_repack<<< 9, 256, 0, stream>>>(enc_w4, (_Float16*)(ws + oPe4),  9,  576,  9, of3);
    k_repack<<< 9, 256, 0, stream>>>(dec_w1, (_Float16*)(ws + oPd1),  9,  576,  9, of3);

    // conv-transpose parity taps (torch [Cin,Cout,4,4]: coutStride=16, cinStride=1024)
    Taps16 tpt[4]; Offs16 oft[4];
    for (int py = 0; py < 2; ++py)
        for (int px = 0; px < 2; ++px) {
            int par = py * 2 + px;
            tpt[par] = Taps16{}; oft[par] = Offs16{};
            for (int a2 = 0; a2 < 2; ++a2)
                for (int a3 = 0; a3 < 2; ++a3) {
                    int t = a2 * 2 + a3;
                    int ky = py ? (2 * a2) : (1 + 2 * a2);
                    int kx = px ? (2 * a3) : (1 + 2 * a3);
                    tpt[par].dy[t] = py ? (1 - a2) : (-a2);
                    tpt[par].dx[t] = px ? (1 - a3) : (-a3);
                    oft[par].off[t] = ky * 4 + kx;
                }
            k_repack<<<4, 256, 0, stream>>>(dec_tw1, (_Float16*)(ws + oPt1) + par * 16384, 4, 16, 1024, oft[par]);
            k_repack<<<4, 256, 0, stream>>>(dec_tw2, (_Float16*)(ws + oPt2) + par * 16384, 4, 16, 1024, oft[par]);
        }

    auto ilog2 = [](int v) { int s = 0; while ((1 << s) < v) ++s; return s; };

    auto conv = [&](int ntaps, const _Float16* in, int Hin, int Win, size_t pkOff,
                    const float* bias, _Float16* out, int Hout, int Wout, int Hs, int Ws,
                    int ostep, int oy0, int ox0, int istride, const Taps16& tp, int relu) {
        long P = 8L * Hs * Ws;
        int blocks = (int)(P / (16 * 8));   // 16 px/wave, 8 waves/block
        int wS = ilog2(Ws), hS = ilog2(Hs);
        const _Float16* pk = (const _Float16*)(ws + pkOff);
        if (ntaps == 16)
            k_conv_wmma<16><<<blocks, 256, 0, stream>>>(in, Hin, Win, pk, bias, out, Hout, Wout,
                                                        wS, hS, ostep, oy0, ox0, istride, tp, relu);
        else if (ntaps == 9)
            k_conv_wmma<9><<<blocks, 256, 0, stream>>>(in, Hin, Win, pk, bias, out, Hout, Wout,
                                                       wS, hS, ostep, oy0, ox0, istride, tp, relu);
        else
            k_conv_wmma<4><<<blocks, 256, 0, stream>>>(in, Hin, Win, pk, bias, out, Hout, Wout,
                                                       wS, hS, ostep, oy0, ox0, istride, tp, relu);
    };

    // ---- encoder ----
    k_enc1<<<2048, 256, 0, stream>>>(x, enc_w1, enc_b1, A1);                    // -> [8,256,256,64]
    conv(16, A1, 256, 256, oPe2, enc_b2, A2, 128, 128, 128, 128, 1, 0, 0, 2, tp4, 1);
    conv(16, A2, 128, 128, oPe3, enc_b3, A3,  64,  64,  64,  64, 1, 0, 0, 2, tp4, 1);
    conv( 9, A3,  64,  64, oPe4, enc_b4, Z,   64,  64,  64,  64, 1, 0, 0, 1, tp3, 0);  // z (no relu)

    // ---- vector quantizer ----
    k_zero<<<1, 64, 0, stream>>>(acc);
    k_vq<<<128, 256, 0, stream>>>(Z, codebook, ZQ, acc);

    // ---- decoder ----
    conv(9, ZQ, 64, 64, oPd1, dec_b1, D1, 64, 64, 64, 64, 1, 0, 0, 1, tp3, 1);
    for (int par = 0; par < 4; ++par) {
        int py = par >> 1, px = par & 1;
        conv(4, D1, 64, 64, oPt1 + (size_t)par * 32768, dec_tb1,
             DT1, 128, 128, 64, 64, 2, py, px, 1, tpt[par], 1);
    }
    for (int par = 0; par < 4; ++par) {
        int py = par >> 1, px = par & 1;
        conv(4, DT1, 128, 128, oPt2 + (size_t)par * 32768, dec_tb2,
             DT2, 256, 256, 128, 128, 2, py, px, 1, tpt[par], 1);
    }
    k_dect3<<<8192, 256, 0, stream>>>(DT2, dec_tw3, dec_tb3, (float*)d_out);    // -> [8,512,512] + sigmoid

    // ---- loss scalar ----
    k_loss<<<1, 64, 0, stream>>>(acc, (float*)d_out + 2097152);
}
